// cmfsm_81423989997959
// MI455X (gfx1250) — matrix operand order repaired and checked by reading, not compiled, not run on hardware
//
#include <hip/hip_runtime.h>

// MI455X / gfx1250, wave32. Fused correlation-similarity + softmax.
// Kernel 1: Lbuf[b][y][x][32] = W0[:,0:32] @ lr_feature   (WMMA f16)
// Kernel 2: per 16-pixel group: Hh = W0[:,32:64] @ hr (2 WMMAs), then per
//           branch h0 = Hh + Lbuf[neighbor] + dm-term, MLP layers 1/2 as
//           WMMAs, layer 3 as an 8-wide dot, softmax over 9 branches.

typedef __attribute__((ext_vector_type(16))) _Float16 v16h;
typedef __attribute__((ext_vector_type(8)))  float    v8f;

#define SB 4
#define SC 32
#define HL 64
#define WL 192
#define SS 4
#define HH 256
#define WW 768

__device__ __forceinline__ float lrelu(float x) { return x > 0.f ? x : 0.01f * x; }

// swap lane halves (lane <-> lane^16) via LDS swizzle hardware (no LDS memory)
__device__ __forceinline__ float swap16(float x) {
  return __int_as_float(__builtin_amdgcn_ds_swizzle(__float_as_int(x), 0x401F));
}

// xvals = {-2,-1,1,2} without a memory-indexed table
__device__ __forceinline__ float xvf(int i) { return (float)(i < 2 ? i - 2 : i - 1); }

// Build a 16x32 f16 A-fragment from a row-major f32 matrix (ld = row stride).
// Layout: lane m=lane&15; element i -> K=i+kh (contiguous-16),
// kh = (lane<16 ? 0 : 16). Zero-pads outside (mvalid, kvalid).
__device__ __forceinline__ v16h load_a_frag(const float* __restrict__ Wm, int ld,
                                            int m0, int k0, int mvalid, int kvalid,
                                            int lane) {
  int m  = lane & 15;
  int kh = (lane < 16) ? 0 : 16;
  v16h a;
#pragma unroll
  for (int i = 0; i < 16; ++i) {
    int k = i + kh;
    float v = 0.f;
    if (m < mvalid && k < kvalid) v = Wm[(size_t)(m0 + m) * ld + (k0 + k)];
    a[i] = (_Float16)v;
  }
  return a;
}

__global__ __launch_bounds__(256) void cmfsm_pre(const float* __restrict__ lrf,
                                                 const float* __restrict__ W0,
                                                 float* __restrict__ Lbuf) {
  const int lane = threadIdx.x & 31;
  const int wave = threadIdx.x >> 5;
  const int W16 = WL / 16;                       // 12
  int g = blockIdx.x * 8 + wave;                 // 16-cell group id
  if (g >= SB * HL * W16) return;
  int b  = g / (HL * W16);
  int r0 = g % (HL * W16);
  int y  = r0 / W16;
  int x0 = (r0 % W16) * 16;

  v16h aA0 = load_a_frag(W0, 66, 0,  0, 16, 32, lane);
  v16h aA1 = load_a_frag(W0, 66, 16, 0, 16, 32, lane);

  // B fragment of lr columns: lane<16 -> pixel=lane, K(ch)=0..15;
  // lane>=16 -> pixel=lane-16, K=16..31 (contiguous-16 B layout).
  int p  = lane & 15;
  int kb = (lane < 16) ? 0 : 16;
  v16h bl;
  const float* lp = lrf + (((size_t)b * SC + kb) * HL + y) * WL + (x0 + p);
#pragma unroll
  for (int i = 0; i < 16; ++i) bl[i] = (_Float16)lp[(size_t)i * (HL * WL)];

  v8f zero = {};
  v8f D0 = __builtin_amdgcn_wmma_f32_16x16x32_f16(false, aA0, false, bl, (short)0, zero, false, false);
  v8f D1 = __builtin_amdgcn_wmma_f32_16x16x32_f16(false, aA1, false, bl, (short)0, zero, false, false);

  // D layout: lane<16 holds ch 0..7 (D0) / 16..23 (D1); lane>=16 holds +8.
  int cb = (lane < 16) ? 0 : 8;
  float* op = Lbuf + ((((size_t)b * HL + y) * WL) + (x0 + p)) * 32;
#pragma unroll
  for (int r = 0; r < 8; ++r) {
    op[cb + r]      = D0[r];
    op[cb + 16 + r] = D1[r];
  }
}

__global__ __launch_bounds__(256) void cmfsm_main(const float* __restrict__ hrf,
                                                  const float* __restrict__ W0,
                                                  const float* __restrict__ W1,
                                                  const float* __restrict__ W2,
                                                  const float* __restrict__ W3,
                                                  const float* __restrict__ Lbuf,
                                                  float* __restrict__ out) {
  const int lane = threadIdx.x & 31;
  const int wave = threadIdx.x >> 5;
  const int W16 = WW / 16;                       // 48
  int g = blockIdx.x * 8 + wave;
  if (g >= SB * HH * W16) return;
  int b  = g / (HH * W16);
  int r0 = g % (HH * W16);
  int h  = r0 / W16;
  int w0 = (r0 % W16) * 16;

  // Weight fragments (built once per wave)
  v16h aW0b0 = load_a_frag(W0, 66, 0,  32, 16, 32, lane);   // W0 cols 32..63, rows 0..15
  v16h aW0b1 = load_a_frag(W0, 66, 16, 32, 16, 32, lane);   // rows 16..31
  v16h aW1   = load_a_frag(W1, 32, 0, 0, 16, 32, lane);
  v16h aW2   = load_a_frag(W2, 16, 0, 0, 8, 16, lane);      // zero-padded to 16x32
  float w3r[8];
#pragma unroll
  for (int r = 0; r < 8; ++r) w3r[r] = W3[r];

  // dm weight columns (W0[:,64] and W0[:,65]) for this lane's channel sets
  int cb = (lane < 16) ? 0 : 8;
  float wa64[8], wa65[8], wb64[8], wb65[8];
#pragma unroll
  for (int r = 0; r < 8; ++r) {
    wa64[r] = W0[(size_t)(cb + r) * 66 + 64];
    wa65[r] = W0[(size_t)(cb + r) * 66 + 65];
    wb64[r] = W0[(size_t)(cb + 16 + r) * 66 + 64];
    wb65[r] = W0[(size_t)(cb + 16 + r) * 66 + 65];
  }

  // hr B fragment (hr is read exactly once over the whole dispatch)
  int p  = lane & 15;
  int kb = (lane < 16) ? 0 : 16;
  v16h bhr;
  {
    const float* hp = hrf + (((size_t)b * SC + kb) * HH + h) * WW + (w0 + p);
#pragma unroll
    for (int i = 0; i < 16; ++i) bhr[i] = (_Float16)hp[(size_t)i * (HH * WW)];
  }
  v8f zero = {};
  v8f D0 = __builtin_amdgcn_wmma_f32_16x16x32_f16(false, aW0b0, false, bhr, (short)0, zero, false, false);
  v8f D1 = __builtin_amdgcn_wmma_f32_16x16x32_f16(false, aW0b1, false, bhr, (short)0, zero, false, false);

  // dm scalars per branch, depend only on (h%4, w%4)
  int rr = h & 3, cc = (w0 + p) & 3;
  float d0s[9], d1s[9];
  d0s[0] = xvf(cc);            d1s[0] = xvf(rr);            // dm
  d0s[1] = (float)(4 - cc);    d1s[1] = xvf(rr);            // dm1
  d0s[2] = (float)(cc + 1);    d1s[2] = xvf(rr);            // dm2
  d0s[3] = xvf(cc);            d1s[3] = (float)(4 - rr);    // dm3
  d0s[4] = xvf(cc);            d1s[4] = (float)(rr + 1);    // dm4
  d0s[5] = d0s[1]; d1s[5] = d1s[1];                          // wlt -> dm1
  d0s[6] = d0s[2]; d1s[6] = d1s[2];                          // wrt -> dm2
  d0s[7] = d0s[3]; d1s[7] = d1s[3];                          // wlb -> dm3
  d0s[8] = d0s[4]; d1s[8] = d1s[4];                          // wrb -> dm4
  const int dys[9] = {0, 0, 0, -SS, SS, -SS, -SS, SS, SS};
  const int dxs[9] = {0, -SS, SS, 0, 0, -SS, SS, -SS, SS};

  int cy = h >> 2;
  float ov[9];
#pragma unroll
  for (int br = 0; br < 9; ++br) {
    int yS = h + dys[br];
    int xS = w0 + p + dxs[br];
    bool ok = (yS >= 0) && (yS < HH) && (xS >= 0) && (xS < WW);
    int cyn = cy + dys[br] / SS;
    int cxn = ((w0 + p) >> 2) + dxs[br] / SS;
    cyn = cyn < 0 ? 0 : (cyn > HL - 1 ? HL - 1 : cyn);
    cxn = cxn < 0 ? 0 : (cxn > WL - 1 ? WL - 1 : cxn);
    const float* Lp = Lbuf + ((((size_t)b * HL + cyn) * WL) + cxn) * 32;

    float t0[8], t1[8];
#pragma unroll
    for (int r = 0; r < 8; ++r) {
      t0[r] = lrelu(D0[r] + Lp[cb + r]      + wa64[r] * d0s[br] + wa65[r] * d1s[br]);
      t1[r] = lrelu(D1[r] + Lp[cb + 16 + r] + wb64[r] * d0s[br] + wb65[r] * d1s[br]);
    }
    // repack h0 (f32 D-layout) -> f16 B-fragment via lane-half swap
    v16h b0;
#pragma unroll
    for (int r = 0; r < 8; ++r) {
      float s0 = swap16(t0[r]);
      float s1 = swap16(t1[r]);
      b0[r]     = (_Float16)((lane < 16) ? t0[r] : s1);
      b0[8 + r] = (_Float16)((lane < 16) ? s0    : t1[r]);
    }
    v8f H1 = __builtin_amdgcn_wmma_f32_16x16x32_f16(false, aW1, false, b0, (short)0, zero, false, false);

    v16h b1v;
#pragma unroll
    for (int r = 0; r < 8; ++r) {
      float u = lrelu(H1[r]);
      float s = swap16(u);
      b1v[r]     = (_Float16)((lane < 16) ? u : 0.f);
      b1v[8 + r] = (_Float16)((lane < 16) ? s : 0.f);
    }
    v8f H2 = __builtin_amdgcn_wmma_f32_16x16x32_f16(false, aW2, false, b1v, (short)0, zero, false, false);

    float o = 0.f;
#pragma unroll
    for (int r = 0; r < 8; ++r) o += w3r[r] * lrelu(H2[r]);
    ov[br] = ok ? o : -100.f;
  }

  // softmax over the 9 branch logits (all in-lane)
  float m = ov[0];
#pragma unroll
  for (int i = 1; i < 9; ++i) m = fmaxf(m, ov[i]);
  float s = 0.f;
  float e[9];
#pragma unroll
  for (int i = 0; i < 9; ++i) { e[i] = __expf(ov[i] - m); s += e[i]; }
  float inv = 1.f / s;
  if (lane < 16) {
#pragma unroll
    for (int br = 0; br < 9; ++br)
      out[(((size_t)b * 9 + br) * HH + h) * WW + (w0 + p)] = e[br] * inv;
  }
}

extern "C" void kernel_launch(void* const* d_in, const int* in_sizes, int n_in,
                              void* d_out, int out_size, void* d_ws, size_t ws_size,
                              hipStream_t stream) {
  (void)in_sizes; (void)n_in; (void)out_size; (void)ws_size;
  const float* lr = (const float*)d_in[0];
  const float* hr = (const float*)d_in[1];
  // d_in[2], d_in[3] (lr_feature_r / hr_feature_r) are unused by the reference
  const float* W0 = (const float*)d_in[4];
  const float* W1 = (const float*)d_in[5];
  const float* W2 = (const float*)d_in[6];
  const float* W3 = (const float*)d_in[7];
  float* out  = (float*)d_out;
  float* Lbuf = (float*)d_ws;   // SB*HL*WL*32 floats = 6.3 MB

  int ngPre  = SB * HL * (WL / 16);   // 3072 waves
  int ngMain = SB * HH * (WW / 16);   // 49152 waves
  cmfsm_pre<<<ngPre / 8, 256, 0, stream>>>(lr, W0, Lbuf);
  cmfsm_main<<<ngMain / 8, 256, 0, stream>>>(hr, W0, W1, W2, W3, Lbuf, out);
}
